// IPDDP_20976620274164
// MI455X (gfx1250) — matrix operand order repaired and checked by reading, not compiled, not run on hardware
//
#include <hip/hip_runtime.h>
#include <stdint.h>

#define T_ 64
#define B_ 128
#define N_ 16
#define M_ 8
#define D_ 24
#define NC_ 8
#define MU_ 0.1f
#define REG_ 1e-6f

typedef __attribute__((ext_vector_type(16))) _Float16 v16h;
typedef __attribute__((ext_vector_type(8)))  _Float16 v8h;
typedef __attribute__((ext_vector_type(8)))  float    v8f;
typedef unsigned int u32x4 __attribute__((ext_vector_type(4)));
typedef int          i32x4 __attribute__((ext_vector_type(4)));
typedef int          i32x8 __attribute__((ext_vector_type(8)));

#if defined(__has_builtin)
#if __has_builtin(__builtin_amdgcn_tensor_load_to_lds)
#define HAVE_TDM 1
#endif
#endif

// ---------- CDNA5 wave32 WMMA fragment index maps (CDNA5 ISA §7.12.2) ----------
// A 16x32 f16: lanes 0-15 hold M=lane, K 0..15; lanes 16-31: M=lane-16, K 16..31 (K = 16h + e, this code's order).
static __device__ inline void a_idx(int lane, int e, int& row, int& k) {
  const int h = (lane >> 4) & 1;
  row = lane & 15;
  k = 16 * h + e;
}
// B 32x16 f16: lane col = lane&15; K = 16h + e (this code's order, not the ISA table).
// C/D f32 16x16: VGPR r holds row r+8h, col = lane&15.

static __device__ inline v8f load_c_f32(const float* p, int ld, int lane) {
  const int h = (lane >> 4) & 1;
  const int col = lane & 15;
  v8f c;
#pragma unroll
  for (int r = 0; r < 8; ++r) c[r] = p[(r + 8 * h) * ld + col];
  return c;
}
static __device__ inline void store_d_f32(float* p, int ld, v8f acc, int lane) {
  const int h = (lane >> 4) & 1;
  const int col = lane & 15;
#pragma unroll
  for (int r = 0; r < 8; ++r) p[(r + 8 * h) * ld + col] = acc[r];
}
// Unconditional accumulate into a 32x32-padded f32 tile (pads receive exact zeros).
static __device__ inline void store_d_f32_add32(float* p, int r0, int n0, v8f acc, int lane) {
  const int h = (lane >> 4) & 1;
  const int col = n0 + (lane & 15);
#pragma unroll
  for (int r = 0; r < 8; ++r) {
    const int row = r0 + r + 8 * h;
    p[row * 32 + col] += acc[r];
  }
}
// Store D into a B-fragment-ordered f16 tile: lane's 8 halves are contiguous -> one b128 store.
static __device__ inline void store_d_fragB(_Float16* ftile, v8f acc, int lane) {
  const int h = (lane >> 4) & 1;
  const int col = lane & 15;
  v8h pk;
#pragma unroll
  for (int r = 0; r < 8; ++r) pk[r] = (_Float16)acc[r];
  *(v8h*)&ftile[col * 16 + 8 * h] = pk;
}

// ---------- Tensor Data Mover: async 1-D f32 tile HBM -> LDS (CDNA5 ISA) ----------
static __device__ inline void tdm_load_f32(const float* gsrc, float* lds_dst, unsigned int elems) {
#ifdef HAVE_TDM
  const unsigned long long ga = (unsigned long long)(uintptr_t)gsrc;
  u32x4 g0;
  g0[0] = 1u;                                                   // count=1, user mode, no gather
  g0[1] = (unsigned int)(uintptr_t)lds_dst;                     // lds_addr (LDS byte offset)
  g0[2] = (unsigned int)ga;                                     // global_addr[31:0]
  g0[3] = (unsigned int)((ga >> 32) & 0x1FFFFFFu) | (2u << 30); // global_addr[56:32] | type=2
  i32x8 g1;
  g1[0] = (int)(2u << 16);                                      // wg_mask=0, data_size=4B
  g1[1] = (int)((elems & 0xFFFFu) << 16);                       // tensor_dim0[15:0]
  g1[2] = (int)((elems >> 16) | (1u << 16));                    // tensor_dim0[31:16], tensor_dim1=1
  g1[3] = (int)(elems << 16);                                   // tile_dim0 = elems
  g1[4] = 1;                                                    // tile_dim1=1
  g1[5] = (int)elems;                                           // tensor_dim0_stride[31:0]
  g1[6] = 0;
  g1[7] = 0;
  const i32x4 z4 = {0, 0, 0, 0};
#if __clang_major__ >= 23
  const i32x8 z8 = {0, 0, 0, 0, 0, 0, 0, 0};
  __builtin_amdgcn_tensor_load_to_lds(g0, g1, z4, z4, z8, 0);
#else
  __builtin_amdgcn_tensor_load_to_lds(g0, g1, z4, z4, 0);
#endif
#else
  (void)gsrc; (void)lds_dst; (void)elems;
#endif
}

// ---------- backward Riccati scan: one workgroup (2 waves) per batch element ----------

__global__ __launch_bounds__(64) void ipddp_bwd(
    const float* __restrict__ Q, const float* __restrict__ q,
    const float* __restrict__ F, const float* __restrict__ G,
    const float* __restrict__ W, const float* __restrict__ cvec,
    const float* __restrict__ s, const float* __restrict__ VT,
    const float* __restrict__ vT, float* __restrict__ Kall,
    float* __restrict__ kall)
{
  const int b    = blockIdx.x;
  const int tid  = threadIdx.x;
  const int lane = tid & 31;
  const bool w0  = (tid < 32);

  __shared__ __align__(16) float sVp[N_ * 32];    // V, zero-padded cols 16..31
  __shared__ __align__(16) float sVt[N_ * N_];
  __shared__ float sv[N_], svt[N_];
  __shared__ __align__(16) float sQh[32 * 32];    // Qh in 24x24 region, ld=32 (pads written w/ zeros)
  __shared__ float sqh[D_];
  __shared__ __align__(16) float sUp[32 * 32];    // [F(16x24); W(8x24); 0], zero-padded, ld=32
  __shared__ float s_sc[NC_], s_smr[NC_];
  __shared__ float sK[M_ * N_], sk[M_], sE[M_ * N_], stq[M_];
  __shared__ float sAug[M_ * 26], sfac[M_];
  __shared__ __align__(32) _Float16 fZ[2][512];   // Z = [VF ; -sc*W ; 0], B-fragment order per tile
#ifdef HAVE_TDM
  __shared__ __align__(16) float sG[N_ * D_ * D_];  // 36 KB G tile, filled by TDM
#endif

#ifdef HAVE_TDM
  if (w0)   // kick off G tile for t = T-1 while we set up
    tdm_load_f32(G + ((size_t)(T_ - 1) * B_ + b) * (size_t)(N_ * D_ * D_), sG, N_ * D_ * D_);
#endif

  // persistent zero pads
  for (int i = tid; i < 32 * 32; i += 64) sUp[i] = 0.f;
  for (int i = tid; i < N_ * 32; i += 64) sVp[i] = 0.f;
  // init value function
  for (int i = tid; i < N_ * N_; i += 64) {
    const int r = i >> 4, c = i & 15;
    sVp[r * 32 + c] = VT[(size_t)b * N_ * N_ + i];
  }
  if (tid < N_) sv[tid] = vT[(size_t)b * N_ + tid];
  __syncthreads();

  for (int t = T_ - 1; t >= 0; --t) {
    const size_t tb = (size_t)t * B_ + b;
    const float* Qt = Q + tb * (D_ * D_);
    const float* qt = q + tb * D_;
    const float* Ft = F + tb * (N_ * D_);
    const float* Gt = G + tb * (size_t)(N_ * D_ * D_);
    const float* Wt = W + tb * (NC_ * D_);
    const float* ct = cvec + tb * NC_;
    const float* st = s + tb * NC_;

    if (t > 0) {
      const size_t tb2 = tb - B_;
      __builtin_prefetch(Q + tb2 * (D_ * D_) + tid * 16, 0, 1);
      __builtin_prefetch(F + tb2 * (N_ * D_), 0, 1);
#ifndef HAVE_TDM
      __builtin_prefetch(G + tb2 * (size_t)(N_ * D_ * D_) + tid * 144, 0, 1);
#endif
    }

    // ---- load step data, float4-vectorized (both waves) ----
    for (int i4 = tid; i4 < 144; i4 += 64) {               // Qt -> sQh (24x24 into ld=32)
      const int row = i4 / 6, c4 = (i4 % 6) * 4;
      *(float4*)&sQh[row * 32 + c4] = *(const float4*)&Qt[row * 24 + c4];
    }
    for (int i4 = tid; i4 < 96; i4 += 64) {                // F -> sUp rows 0..15
      const int row = i4 / 6, c4 = (i4 % 6) * 4;
      *(float4*)&sUp[row * 32 + c4] = *(const float4*)&Ft[row * 24 + c4];
    }
    for (int i4 = tid; i4 < 48; i4 += 64) {                // W -> sUp rows 16..23
      const int row = i4 / 6, c4 = (i4 % 6) * 4;
      *(float4*)&sUp[(16 + row) * 32 + c4] = *(const float4*)&Wt[row * 24 + c4];
    }
    if (tid < D_) sqh[tid] = qt[tid];
    if (tid < NC_) {
      const float c  = ct[tid], sl = st[tid];
      const float r  = sl * c + MU_;
      s_sc[tid]  = sl / c;
      s_smr[tid] = sl - r / c;
    }
#ifdef HAVE_TDM
    if (w0) __builtin_amdgcn_s_wait_tensorcnt(0);          // G[t] resident in sG
#endif
    __syncthreads();

    // ---- qh = qt + F^T v + W^T (s - r/c) ----
    if (tid < D_) {
      float acc = sqh[tid];
      for (int n = 0; n < N_; ++n)  acc += sUp[n * 32 + tid] * sv[n];
      for (int c = 0; c < NC_; ++c) acc += sUp[(16 + c) * 32 + tid] * s_smr[c];
      sqh[tid] = acc;
    }

    // ---- Qh += sum_n v_n * G_n  (float4, 16-way reduction) ----
    {
      float vreg[N_];
#pragma unroll
      for (int n = 0; n < N_; ++n) vreg[n] = sv[n];
#ifdef HAVE_TDM
      const float* gsrc = sG;
#else
      const float* gsrc = Gt;
#endif
      for (int i4 = tid; i4 < 144; i4 += 64) {
        const int row = i4 / 6, c4 = (i4 % 6) * 4;
        float4 acc = *(float4*)&sQh[row * 32 + c4];
        const float* gp = gsrc + row * 24 + c4;
#pragma unroll
        for (int n = 0; n < N_; ++n) {
          const float4 g = *(const float4*)&gp[n * (D_ * D_)];
          acc.x += vreg[n] * g.x; acc.y += vreg[n] * g.y;
          acc.z += vreg[n] * g.z; acc.w += vreg[n] * g.w;
        }
        *(float4*)&sQh[row * 32 + c4] = acc;
      }
    }
    __syncthreads();   // sG fully consumed; sQh base complete

#ifdef HAVE_TDM
    if (w0 && t > 0)   // stream next step's G tile under this step's algebra
      tdm_load_f32(Gt - (size_t)B_ * (N_ * D_ * D_), sG, N_ * D_ * D_);
#endif

    // ---- fill Z fragment rows K=16..31 = [-sc*W ; 0]  (branchless; sUp pads give col zeros) ----
    for (int i = tid; i < 512; i += 64) {
      const int nt = i >> 8, r = i & 255;
      const int lane2 = 16 + (r >> 4), e = r & 15;
      const int col = nt * 16 + (lane2 & 15);
      const int cc = (e < 8) ? e : 0;
      const float x = -s_sc[cc] * sUp[(16 + cc) * 32 + col];
      fZ[nt][lane2 * 16 + e] = (_Float16)((e < 8) ? x : 0.f);
    }

    // ---- WMMA pass 1: VF = V * F  -> fZ rows 0..15 (fragment-order b128 store) ----
    if (w0) {
      v16h a;
#pragma unroll
      for (int e = 0; e < 16; ++e) {
        int row, k; a_idx(lane, e, row, k);
        a[e] = (_Float16)sVp[row * 32 + k];                // unconditional (zero pad)
      }
      const int h = (lane >> 4) & 1, bcol = lane & 15;
#pragma unroll
      for (int nt = 0; nt < 2; ++nt) {
        v16h bb;
#pragma unroll
        for (int e = 0; e < 16; ++e) {
          const int krow = 16 * h + e;
          const int kr = (krow < N_) ? krow : 0;
          const float x = sUp[kr * 32 + nt * 16 + bcol];   // F rows; col pads are zero
          bb[e] = (_Float16)((krow < N_) ? x : 0.f);
        }
        v8f cz = {};
        v8f d = __builtin_amdgcn_wmma_f32_16x16x32_f16(false, a, false, bb, (short)0, cz, false, false);
        store_d_fragB(fZ[nt], d, lane);
      }
    }
    __syncthreads();

    // ---- WMMA pass 2: Qh += U^T * Z  (F^T V F - W^T diag(sc) W fused, K=24) ----
    if (w0) {
      const v16h b0 = *(const v16h*)&fZ[0][lane * 16];     // 2x ds_load_b128 each
      const v16h b1 = *(const v16h*)&fZ[1][lane * 16];
#pragma unroll
      for (int mt = 0; mt < 2; ++mt) {
        v16h a;
#pragma unroll
        for (int e = 0; e < 16; ++e) {
          int row, n; a_idx(lane, e, row, n);
          a[e] = (_Float16)sUp[n * 32 + (mt * 16 + row)];  // single unconditional load
        }
        v8f cz0 = {}, cz1 = {};
        v8f d0 = __builtin_amdgcn_wmma_f32_16x16x32_f16(false, a, false, b0, (short)0, cz0, false, false);
        store_d_f32_add32(sQh, mt * 16, 0, d0, lane);
        v8f d1 = __builtin_amdgcn_wmma_f32_16x16x32_f16(false, a, false, b1, (short)0, cz1, false, false);
        store_d_f32_add32(sQh, mt * 16, 16, d1, lane);
      }
    }
    __syncthreads();

    // ---- 8x8 SPD solve, 17 RHS: branchless Gauss-Jordan on [Quu+REG*I | Qux | qu] ----
    for (int idx = tid; idx < M_ * 25; idx += 64) {
      const int i = idx / 25, j = idx % 25;
      const float v1 = sQh[(N_ + i) * 32 + N_ + ((j < M_) ? j : 0)] + ((i == j) ? REG_ : 0.f);
      const float v2 = sQh[(N_ + i) * 32 + ((j >= M_ && j < 24) ? (j - M_) : 0)];
      const float v3 = sqh[N_ + i];
      sAug[i * 26 + j] = (j < M_) ? v1 : ((j < 24) ? v2 : v3);
    }
    __syncthreads();
    for (int kk = 0; kk < M_; ++kk) {
      if (tid < M_) {
        const float pivinv = 1.f / sAug[kk * 26 + kk];
        const float f = sAug[tid * 26 + kk] * pivinv;
        sfac[tid] = (tid == kk) ? 0.f : f;
      }
      __syncthreads();
      for (int idx = tid; idx < M_ * 25; idx += 64) {
        const int i = idx / 25, j = idx % 25;
        sAug[i * 26 + j] -= sfac[i] * sAug[kk * 26 + j];   // sfac[kk]==0 keeps pivot row
      }
      __syncthreads();
    }
    for (int idx = tid; idx < M_ * N_; idx += 64) {
      const int i = idx / N_, n = idx % N_;
      sK[idx] = -sAug[i * 26 + M_ + n] / sAug[i * 26 + i];
    }
    if (tid < M_) sk[tid] = -sAug[tid * 26 + 24] / sAug[tid * 26 + tid];
    __syncthreads();

    // ---- emit gains ----
    for (int idx = tid; idx < M_ * N_; idx += 64) Kall[tb * (M_ * N_) + idx] = sK[idx];
    if (tid < M_) kall[tb * M_ + tid] = sk[tid];

    // ---- E = Qux + Quu*K ; tq = qu + Quu*k ----
    for (int idx = tid; idx < M_ * N_; idx += 64) {
      const int i = idx / N_, n = idx % N_;
      float acc = sQh[(N_ + i) * 32 + n];
      for (int m = 0; m < M_; ++m) acc += sQh[(N_ + i) * 32 + N_ + m] * sK[m * N_ + n];
      sE[idx] = acc;
    }
    if (tid < M_) {
      float acc = sqh[N_ + tid];
      for (int l = 0; l < M_; ++l) acc += sQh[(N_ + tid) * 32 + N_ + l] * sk[l];
      stq[tid] = acc;
    }
    __syncthreads();

    // ---- WMMA pass 3: Vn = Qxx + [K^T|Qxu]*[E;K]  (C carries Qxx; branchless gathers) ----
    if (w0) {
      v16h a;
#pragma unroll
      for (int e = 0; e < 16; ++e) {
        int row, k; a_idx(lane, e, row, k);
        const float xa = sK[((k < M_) ? k : 0) * N_ + row];
        const float xb = sQh[row * 32 + N_ + ((k >= M_ && k < 2 * M_) ? (k - M_) : 0)];
        a[e] = (_Float16)((k < M_) ? xa : ((k < 2 * M_) ? xb : 0.f));
      }
      const int h = (lane >> 4) & 1, bcol = lane & 15;
      v16h bb;
#pragma unroll
      for (int e = 0; e < 16; ++e) {
        const int krow = 16 * h + e;
        const float xa = sE[((krow < M_) ? krow : 0) * N_ + bcol];
        const float xb = sK[((krow >= M_ && krow < 2 * M_) ? (krow - M_) : 0) * N_ + bcol];
        bb[e] = (_Float16)((krow < M_) ? xa : ((krow < 2 * M_) ? xb : 0.f));
      }
      const v8f cf = load_c_f32(sQh, 32, lane);            // Qxx as accumulator
      v8f d = __builtin_amdgcn_wmma_f32_16x16x32_f16(false, a, false, bb, (short)0, cf, false, false);
      store_d_f32(sVt, N_, d, lane);
    }
    // vn = qx + Qxu*k + K^T*(qu + Quu*k)
    if (tid < N_) {
      float acc = sqh[tid];
      for (int m = 0; m < M_; ++m) acc += sQh[tid * 32 + N_ + m] * sk[m];
      for (int m = 0; m < M_; ++m) acc += sK[m * N_ + tid] * stq[m];
      svt[tid] = acc;
    }
    __syncthreads();

    // ---- symmetrize + commit carried state (into padded sVp) ----
    for (int i = tid; i < N_ * N_; i += 64) {
      const int r = i / N_, c2 = i % N_;
      sVp[r * 32 + c2] = 0.5f * (sVt[i] + sVt[c2 * N_ + r]);
    }
    if (tid < N_) sv[tid] = svt[tid];
    __syncthreads();
  }
}

// ---------- forward rollout: du = K dx + k ; dx' = fx dx + fu du ; us = u_nom + du ----------

__global__ __launch_bounds__(32) void ipddp_fwd(
    const float* __restrict__ F, const float* __restrict__ x0,
    const float* __restrict__ x_nom, const float* __restrict__ u_nom,
    const float* __restrict__ Kall, const float* __restrict__ kall,
    float* __restrict__ us)
{
  const int b = blockIdx.x;
  const int lane = threadIdx.x;
  __shared__ float sdx[N_], sdxn[N_], sdu[M_];

  if (lane < N_) sdx[lane] = x0[(size_t)b * N_ + lane] - x_nom[(size_t)b * N_ + lane];
  __syncthreads();

  for (int t = 0; t < T_; ++t) {
    const size_t tb = (size_t)t * B_ + b;
    if (lane < M_) {
      const float* Km = Kall + tb * (M_ * N_) + lane * N_;
      float du = kall[tb * M_ + lane];
      for (int n = 0; n < N_; ++n) du += Km[n] * sdx[n];
      sdu[lane] = du;
      us[tb * M_ + lane] = u_nom[tb * M_ + lane] + du;
    }
    __syncthreads();
    if (lane < N_) {
      const float* Fr = F + tb * (N_ * D_) + lane * D_;
      float acc = 0.f;
      for (int j = 0; j < N_; ++j) acc += Fr[j] * sdx[j];
      for (int m = 0; m < M_; ++m) acc += Fr[N_ + m] * sdu[m];
      sdxn[lane] = acc;
    }
    __syncthreads();
    if (lane < N_) sdx[lane] = sdxn[lane];
    __syncthreads();
  }
}

extern "C" void kernel_launch(void* const* d_in, const int* in_sizes, int n_in,
                              void* d_out, int out_size, void* d_ws, size_t ws_size,
                              hipStream_t stream) {
  (void)in_sizes; (void)n_in; (void)out_size; (void)ws_size;
  const float* Q     = (const float*)d_in[0];
  const float* q     = (const float*)d_in[1];
  const float* F     = (const float*)d_in[2];
  const float* G     = (const float*)d_in[3];
  const float* W     = (const float*)d_in[4];
  const float* cvec  = (const float*)d_in[5];
  const float* s     = (const float*)d_in[6];
  const float* VT    = (const float*)d_in[7];
  const float* vT    = (const float*)d_in[8];
  const float* x0    = (const float*)d_in[9];
  const float* x_nom = (const float*)d_in[10];
  const float* u_nom = (const float*)d_in[11];
  float* us   = (float*)d_out;
  float* Kall = (float*)d_ws;                               // T*B*M*N floats
  float* kall = Kall + (size_t)T_ * B_ * M_ * N_;           // T*B*M floats

  ipddp_bwd<<<B_, 64, 0, stream>>>(Q, q, F, G, W, cvec, s, VT, vT, Kall, kall);
  ipddp_fwd<<<B_, 32, 0, stream>>>(F, x0, x_nom, u_nom, Kall, kall, us);
}
